// RNN_10634339025509
// MI455X (gfx1250) — matrix operand order repaired and checked by reading, not compiled, not run on hardware
//
#include <hip/hip_runtime.h>
#include <hip/hip_bf16.h>

typedef __bf16 bf16;
typedef __attribute__((ext_vector_type(16))) __bf16 v16bf;
typedef __attribute__((ext_vector_type(8)))  __bf16 v8bf;
typedef __attribute__((ext_vector_type(8)))  float  v8f;

#define T_STEPS 1000
#define BATCH   32
#define DXDIM   720
#define HID     1024
#define OUTD    80
#define DPAD    736                  // DX padded to multiple of 32 (K for WMMA)
#define G3      3072                 // 3*H
#define ROWS    (T_STEPS*BATCH)      // 32000
#define NWG_REC 64

// ---------------------------------------------------------------------------
// Core WMMA micro-kernel: one 16x16 f32 tile, A [16 x K] row-major (lda),
// B = W rows as columns: W row n = output column n, [16 x K] row-major (ldb).
// bf16 data, f32 accumulate, K multiple of 32.
// ---------------------------------------------------------------------------
__device__ __forceinline__ v8f wmma_tile(const bf16* __restrict__ A, int lda,
                                         const bf16* __restrict__ B, int ldb,
                                         int K, v8f acc) {
  const int lane = threadIdx.x & 31;
  const int half = lane >> 4;    // 0: lanes 0-15, 1: lanes 16-31
  const int l16  = lane & 15;
  const bf16* ap = A + (size_t)l16 * lda;
  const bf16* bp = B + (size_t)l16 * ldb;
#pragma unroll 4
  for (int kk = 0; kk < K; kk += 32) {
    // A layout (16-bit, 16x32): lane half 0 -> K {kk..kk+7, kk+16..kk+23},
    //                           half 1 -> K {kk+8..kk+15, kk+24..kk+31}
    v8bf a0 = *(const v8bf*)(ap + kk + half * 8);
    v8bf a1 = *(const v8bf*)(ap + kk + half * 8 + 16);
    // B layout (32x16): lane = column, half 0 -> K {kk..kk+7, kk+16..kk+23}, half 1 -> K {kk+8..kk+15, kk+24..kk+31}
    v8bf b0 = *(const v8bf*)(bp + kk + half * 8);
    v8bf b1 = *(const v8bf*)(bp + kk + half * 8 + 16);
    v16bf a, b;
#pragma unroll
    for (int i = 0; i < 8; ++i) {
      a[i] = a0[i]; a[8 + i] = a1[i];
      b[i] = b0[i]; b[8 + i] = b1[i];
    }
    acc = __builtin_amdgcn_wmma_f32_16x16x32_bf16(
        false, a, false, b, (short)0, acc, false, false);
  }
  return acc;
}

// ---------------------------------------------------------------------------
// Weight prep: W_rec [4096 x 1024] bf16, 4 gate blocks of 1024 rows:
//   [0,1024)   r : W_hh + W_comb          (W_comb = W_ih[:, :80] @ W_ar)
//   [1024,2048) z : W_hh + W_comb
//   [2048,3072) n_h: W_hh                  (scaled by r at use)
//   [3072,4096) n_i: W_comb                (y-feedback part of gi_n)
// ---------------------------------------------------------------------------
__device__ __forceinline__ float comb_elem(const float* W_ih, const float* W_ar,
                                           int g, int k) {
  float s = 0.f;
#pragma unroll 4
  for (int o = 0; o < OUTD; ++o)
    s += W_ih[(size_t)g * 800 + o] * W_ar[(size_t)o * HID + k];
  return s;
}

__global__ void k_prep_wrec(const float* __restrict__ W_ih,
                            const float* __restrict__ W_hh,
                            const float* __restrict__ W_ar,
                            bf16* __restrict__ Wrec) {
  size_t idx = (size_t)blockIdx.x * blockDim.x + threadIdx.x;  // 4096*1024
  int r4 = (int)(idx >> 10);
  int k  = (int)(idx & 1023);
  float v;
  if (r4 < 2048) {                        // r,z folded
    v = W_hh[(size_t)r4 * HID + k] + comb_elem(W_ih, W_ar, r4, k);
  } else if (r4 < 3072) {                 // n_h
    v = W_hh[(size_t)r4 * HID + k];
  } else {                                // n_i
    v = comb_elem(W_ih, W_ar, r4 - 1024, k);
  }
  Wrec[idx] = (bf16)v;
}

__global__ void k_prep_bias(const float* __restrict__ W_ih,
                            const float* __restrict__ b_ih,
                            const float* __restrict__ b_hh,
                            const float* __restrict__ b_ar,
                            float* __restrict__ bias4) {
  int r4 = blockIdx.x * blockDim.x + threadIdx.x;   // 4096
  float v;
  if (r4 < 2048) {
    float s = b_ih[r4] + b_hh[r4];
    for (int o = 0; o < OUTD; ++o) s += W_ih[(size_t)r4 * 800 + o] * b_ar[o];
    v = s;
  } else if (r4 < 3072) {
    v = b_hh[r4];
  } else {
    int g = r4 - 1024;
    float s = b_ih[g];
    for (int o = 0; o < OUTD; ++o) s += W_ih[(size_t)g * 800 + o] * b_ar[o];
    v = s;
  }
  bias4[r4] = v;
}

__global__ void k_conv_wihx(const float* __restrict__ W_ih, bf16* __restrict__ Wx) {
  size_t idx = (size_t)blockIdx.x * blockDim.x + threadIdx.x;  // 3072*736
  int g = (int)(idx / DPAD), k = (int)(idx % DPAD);
  Wx[idx] = (k < DXDIM) ? (bf16)W_ih[(size_t)g * 800 + OUTD + k] : (bf16)0.f;
}

__global__ void k_conv_x(const float* __restrict__ x, bf16* __restrict__ xb) {
  size_t idx = (size_t)blockIdx.x * blockDim.x + threadIdx.x;  // 32000*736
  size_t row = idx / DPAD;
  int k = (int)(idx % DPAD);
  xb[idx] = (k < DXDIM) ? (bf16)x[row * DXDIM + k] : (bf16)0.f;
}

__global__ void k_conv_war(const float* __restrict__ W_ar, bf16* __restrict__ w) {
  size_t idx = (size_t)blockIdx.x * blockDim.x + threadIdx.x;  // 80*1024
  w[idx] = (bf16)W_ar[idx];
}

__global__ void k_init(bf16* __restrict__ h_hist, unsigned* __restrict__ cnt) {
  size_t idx = (size_t)blockIdx.x * blockDim.x + threadIdx.x;  // 32*1024
  h_hist[idx] = (bf16)0.f;   // h0 = 0 (slot 0)
  if (idx == 0) *cnt = 0u;
}

// ---------------------------------------------------------------------------
// gi_x[32000, 3072] = x_bf16[32000, 736] @ W_ihx[3072, 736].T   (f32 out)
// One wave per 16x16 tile: 2000 row-tiles x 192 col-tiles = 384000 waves.
// ---------------------------------------------------------------------------
__global__ __launch_bounds__(256) void k_gix(const bf16* __restrict__ xb,
                                             const bf16* __restrict__ wx,
                                             float* __restrict__ gix) {
  unsigned wave = blockIdx.x * 8u + (threadIdx.x >> 5);
  unsigned rt = wave / 192u, ct = wave % 192u;
  const int lane = threadIdx.x & 31, half = lane >> 4, l16 = lane & 15;
  v8f acc = {};
  acc = wmma_tile(xb + (size_t)(rt * 16) * DPAD, DPAD,
                  wx + (size_t)(ct * 16) * DPAD, DPAD, DPAD, acc);
  // C layout: VGPR i, lanes<16 -> M=i ; lanes>=16 -> M=8+i ; N=l16
#pragma unroll
  for (int i = 0; i < 8; ++i) {
    size_t row = (size_t)rt * 16 + i + half * 8;
    gix[row * G3 + ct * 16 + l16] = acc[i];
  }
}

// ---------------------------------------------------------------------------
// Persistent recurrent kernel. 64 WGs x 256 threads. WG owns 16 hidden units.
// Wave w: gate g = w>>1 (0=r,1=z,2=n_h,3=n_i), batch row-tile m = w&1.
// Per step: 8 WMMA tiles [16x16], K=1024; LDS gate fusion; grid barrier.
// ---------------------------------------------------------------------------
__global__ __launch_bounds__(256, 1) void k_recurrent(
    const bf16*  __restrict__ Wrec,   // [4096,1024]
    const float* __restrict__ bias4,  // [4096]
    const float* __restrict__ gix,    // [32000,3072]
    bf16*        __restrict__ h_hist, // [(T+1)*32, 1024]; slot 0 = h0
    unsigned*    __restrict__ counter) {
  __shared__ float lds[8][8][32];
  const int tid  = threadIdx.x;
  const int w    = tid >> 5;
  const int g    = w >> 1;
  const int m    = w & 1;
  const int lane = tid & 31;
  const int jb   = blockIdx.x * 16;        // hidden-unit block base
  const int cc   = tid & 15;               // combine: column
  const int bb   = tid >> 4;               // combine: batch row base (0..15)

  const bf16* Bp = Wrec + (size_t)(g * HID + jb) * HID;

#pragma unroll 1
  for (int t = 0; t < T_STEPS; ++t) {
    const bf16* Ap = h_hist + (size_t)t * (BATCH * HID) + (size_t)(m * 16) * HID;
    v8f acc = {};
    acc = wmma_tile(Ap, HID, Bp, HID, HID, acc);
#pragma unroll
    for (int i = 0; i < 8; ++i) lds[w][i][lane] = acc[i];
    __syncthreads();

    // gate fusion: element (b, c): m=b/16, brow=b%16, vgpr=brow&7,
    // lane = c + ((brow&8)?16:0)
#pragma unroll
    for (int p = 0; p < 2; ++p) {
      int b  = bb + p * 16;
      int bm = b >> 4, br = b & 15;
      int v  = br & 7, ln = (br & 8) ? (cc + 16) : cc;
      float aR  = lds[0 + bm][v][ln];
      float aZ  = lds[2 + bm][v][ln];
      float aNH = lds[4 + bm][v][ln];
      float aNI = lds[6 + bm][v][ln];
      int j = jb + cc;
      const float* gx = gix + ((size_t)t * BATCH + b) * G3;
      float pr = aR + gx[j]        + bias4[j];
      float pz = aZ + gx[HID + j]  + bias4[HID + j];
      float r  = 1.f / (1.f + __expf(-pr));
      float z  = 1.f / (1.f + __expf(-pz));
      float n  = tanhf(aNI + gx[2 * HID + j] + bias4[3 * HID + j] +
                       r * (aNH + bias4[2 * HID + j]));
      float hp = (float)h_hist[(size_t)t * (BATCH * HID) + (size_t)b * HID + j];
      float hn = (1.f - z) * n + z * hp;
      h_hist[(size_t)(t + 1) * (BATCH * HID) + (size_t)b * HID + j] = (bf16)hn;
    }

    // grid-wide barrier: release (global_wb) -> atomic arrive -> spin -> acquire (global_inv)
    __builtin_amdgcn_fence(__ATOMIC_RELEASE, "agent");
    __syncthreads();
    if (tid == 0) {
      __hip_atomic_fetch_add(counter, 1u, __ATOMIC_RELAXED, __HIP_MEMORY_SCOPE_AGENT);
      unsigned target = (unsigned)NWG_REC * (unsigned)(t + 1);
      while (__hip_atomic_load(counter, __ATOMIC_RELAXED, __HIP_MEMORY_SCOPE_AGENT) < target)
        __builtin_amdgcn_s_sleep(1);
    }
    __syncthreads();
    __builtin_amdgcn_fence(__ATOMIC_ACQUIRE, "agent");
  }
}

// ---------------------------------------------------------------------------
// y[32000, 80] = h[32000, 1024] @ W_ar[80, 1024].T + b_ar
// h rows are h_hist slots 1..T (shift by 32 rows). 2000 x 5 tiles.
// ---------------------------------------------------------------------------
__global__ __launch_bounds__(256) void k_yhead(const bf16* __restrict__ h_hist,
                                               const bf16* __restrict__ war,
                                               const float* __restrict__ b_ar,
                                               float* __restrict__ out) {
  unsigned wave = blockIdx.x * 8u + (threadIdx.x >> 5);
  unsigned rt = wave / 5u, ct = wave % 5u;
  const int lane = threadIdx.x & 31, half = lane >> 4, l16 = lane & 15;
  v8f acc = {};
  acc = wmma_tile(h_hist + (size_t)(BATCH + rt * 16) * HID, HID,
                  war + (size_t)(ct * 16) * HID, HID, HID, acc);
#pragma unroll
  for (int i = 0; i < 8; ++i) {
    size_t row = (size_t)rt * 16 + i + half * 8;
    int col = ct * 16 + l16;
    out[row * OUTD + col] = acc[i] + b_ar[col];
  }
}

// ---------------------------------------------------------------------------
extern "C" void kernel_launch(void* const* d_in, const int* in_sizes, int n_in,
                              void* d_out, int out_size, void* d_ws, size_t ws_size,
                              hipStream_t stream) {
  const float* x    = (const float*)d_in[0];
  const float* W_ih = (const float*)d_in[1];
  const float* W_hh = (const float*)d_in[2];
  const float* b_ih = (const float*)d_in[3];
  const float* b_hh = (const float*)d_in[4];
  const float* W_ar = (const float*)d_in[5];
  const float* b_ar = (const float*)d_in[6];
  float* out = (float*)d_out;

  char* ws = (char*)d_ws;
  size_t off = 0;
  auto take = [&](size_t bytes) {
    size_t o = off;
    off += (bytes + 255) & ~(size_t)255;
    return o;
  };
  bf16*     WREC  = (bf16*)(ws + take((size_t)4096 * HID * 2));
  float*    BIAS4 = (float*)(ws + take((size_t)4096 * 4));
  bf16*     WIHX  = (bf16*)(ws + take((size_t)G3 * DPAD * 2));
  bf16*     XB    = (bf16*)(ws + take((size_t)ROWS * DPAD * 2));
  bf16*     WARB  = (bf16*)(ws + take((size_t)OUTD * HID * 2));
  float*    GIX   = (float*)(ws + take((size_t)ROWS * G3 * 4));
  bf16*     HH    = (bf16*)(ws + take((size_t)(ROWS + BATCH) * HID * 2));
  unsigned* CNT   = (unsigned*)(ws + take(256));

  // weight / input prep (parallel, off critical path)
  k_prep_wrec<<<(4096 * HID) / 256, 256, 0, stream>>>(W_ih, W_hh, W_ar, WREC);
  k_prep_bias<<<16, 256, 0, stream>>>(W_ih, b_ih, b_hh, b_ar, BIAS4);
  k_conv_wihx<<<(G3 * DPAD) / 256, 256, 0, stream>>>(W_ih, WIHX);
  k_conv_x<<<((size_t)ROWS * DPAD) / 256, 256, 0, stream>>>(x, XB);
  k_conv_war<<<(OUTD * HID) / 256, 256, 0, stream>>>(W_ar, WARB);
  k_init<<<(BATCH * HID) / 256, 256, 0, stream>>>(HH, CNT);

  // big parallel GEMM: gi_x for all timesteps (142 GFLOP of WMMA)
  k_gix<<<(2000 * 192) / 8, 256, 0, stream>>>(XB, WIHX, GIX);

  // serial recurrence: one grid sync per step
  k_recurrent<<<NWG_REC, 256, 0, stream>>>(WREC, BIAS4, GIX, HH, CNT);

  // batched output head
  k_yhead<<<(2000 * 5) / 8, 256, 0, stream>>>(HH, WARB, b_ar, out);
}
